// Model_ConvLSTM2D_5669356830866
// MI455X (gfx1250) — matrix operand order repaired and checked by reading, not compiled, not run on hardware
//
#include <hip/hip_runtime.h>
#include <hip/hip_bf16.h>

typedef _Float16 f16;
typedef __attribute__((ext_vector_type(16))) _Float16 v16h;
typedef __attribute__((ext_vector_type(8)))  _Float16 v8h;
typedef __attribute__((ext_vector_type(4)))  _Float16 v4h;
typedef __attribute__((ext_vector_type(8)))  float    v8f;

union V16H { v16h v; v8h h[2]; };

__device__ __forceinline__ float hsig(float x) {
    float y = x * 0.2f + 0.5f;
    return fminf(fmaxf(y, 0.0f), 1.0f);
}

// ---------------------------------------------------------------------------
// f32 -> f16 elementwise convert
// ---------------------------------------------------------------------------
__global__ void k_f32_to_f16(const float* __restrict__ src, f16* __restrict__ dst, int n) {
    int i = blockIdx.x * blockDim.x + threadIdx.x;
    if (i < n) dst[i] = (f16)src[i];
}

// ---------------------------------------------------------------------------
// Weight prep: wA [KAR][N] -> k in [0,KAR); zeros in [KAR,KAP);
// wB [KB][N] -> k in [KAP, KAP+KB); zeros beyond (up to Kp).
// Transposed to [N][Kp] f16 so B fragments are contiguous 16B loads.
// ---------------------------------------------------------------------------
__global__ void k_prep_bt(const float* __restrict__ wA, int KAR,
                          const float* __restrict__ wB, int KAP, int KB,
                          int N, int Kp, f16* __restrict__ bt) {
    int i = blockIdx.x * blockDim.x + threadIdx.x;
    if (i >= N * Kp) return;
    int n = i / Kp, k = i - n * Kp;
    float v = 0.0f;
    if (k < KAR)                          v = wA[k * N + n];
    else if (k >= KAP && k < KAP + KB)    v = wB[(k - KAP) * N + n];
    bt[i] = (f16)v;
}

// Head weights: pack wl (64x7) and wv (64x1) into Bt [16][64] f16 (cols 8..15 = 0)
__global__ void k_prep_head(const float* __restrict__ wl, const float* __restrict__ wv,
                            f16* __restrict__ bt) {
    int i = blockIdx.x * blockDim.x + threadIdx.x;
    if (i >= 16 * 64) return;
    int n = i >> 6, k = i & 63;
    float v = 0.0f;
    if (n < 7)       v = wl[k * 7 + n];
    else if (n == 7) v = wv[k];
    bt[i] = (f16)v;
}

// ---------------------------------------------------------------------------
// Fused ConvLSTM gate GEMM (geometry fully compile-time):
//   z[m][n] = sum_k [ im2col(x_t) ; im2col(h_prev) ][m][k] * Bt[n][k] + bias[n]
// X path: 3x3, stride S, VALID, channels CX, K in [0,KAR) (zeros to KAP).
// H path: 3x3, stride 1, SAME, channels CH, K in [KAP,KTOT) (zeros to KP).
//
// Staging fast path: when the channel count of a region is 64, a 32-wide
// k-chunk sits inside one (dy,dx) tap => 32 contiguous channels = one 64B
// run with a per-row-uniform bounds predicate. 128 threads = 16 rows x 8
// threads, each moving 8B (b64 load -> b64 LDS store). Irregular chunks
// (CX=3 head chunk, CH=16 tail) use the elementwise fallback.
// ---------------------------------------------------------------------------
template<int TT, int CX, int S, int IH, int IW,
         int CH, int HH, int HW, int OH, int OW,
         int N, int KAR, int KAP, int KTOT, int KP, int NT>
__global__ __launch_bounds__(128) void k_convlstm_gemm(
    const f16* __restrict__ srcX, int t,
    const f16* __restrict__ srcH,
    const f16* __restrict__ Bt,
    const float* __restrict__ bias,
    float* __restrict__ z)
{
    constexpr int NKS = KP / 32;
    __shared__ alignas(16) f16 aT[2][16 * 32];

    const int lane = threadIdx.x & 31;
    const int wave = threadIdx.x >> 5;
    const int m0 = blockIdx.x * 16;
    const int n0 = (blockIdx.y * 4 + wave) * (16 * NT);

    // Hoisted per-thread values for the vector staging path (1 row / thread).
    int syv, sxv, bXv, bHv;
    {
        int m   = m0 + (threadIdx.x >> 3);
        int img = m / (OH * OW);
        int rem = m - img * (OH * OW);
        syv = rem / OW;
        sxv = rem - syv * OW;
        bXv = ((img * TT + t) * IH) * (IW * CX);
        bHv = img * (HH * HW * CH);
    }
    // Hoisted per-thread values for the elementwise fallback (4 slots).
    int sy0[4], sx0[4], bX[4], bH[4];
#pragma unroll
    for (int i = 0; i < 4; ++i) {
        int e   = threadIdx.x + i * 128;
        int m   = m0 + (e >> 5);
        int img = m / (OH * OW);
        int rem = m - img * (OH * OW);
        sy0[i] = rem / OW;
        sx0[i] = rem - sy0[i] * OW;
        bX[i]  = ((img * TT + t) * IH) * (IW * CX);
        bH[i]  = img * (HH * HW * CH);
    }

    const int vrow  = threadIdx.x >> 3;     // 0..15
    const int vpart = (threadIdx.x & 7) * 4; // 0,4,..,28 halves

    auto stage = [&](int ks, f16* buf) {
        if (CH == 64 && ks * 32 >= KAP) {
            // ---- vector H-path chunk: one (dy,dx), 32 contiguous channels
            int base = ks * 32 - KAP;           // multiple of 32
            int dy = base / 192;                // uniform per chunk
            int r  = base - dy * 192;
            int dx = r >> 6;
            int cb = r & 63;
            int iy = syv + dy - 1;
            int ix = sxv + dx - 1;
            v4h vals = {};
            if (srcH && iy >= 0 && iy < HH && ix >= 0 && ix < HW)
                vals = *(const v4h*)(srcH + bHv + (iy * HW + ix) * CH + cb + vpart);
            *(v4h*)(buf + vrow * 32 + vpart) = vals;
        } else if (CX == 64 && (ks + 1) * 32 <= KAP) {
            // ---- vector X-path chunk: one (dy,dx), 32 contiguous channels
            int base = ks * 32;                 // multiple of 32
            int dy = base / 192;
            int r  = base - dy * 192;
            int dx = r >> 6;
            int cb = r & 63;
            int iy = syv * S + dy;
            int ix = sxv * S + dx;
            v4h vals = {};
            if (iy < IH && ix < IW)
                vals = *(const v4h*)(srcX + bXv + (iy * IW + ix) * CX + cb + vpart);
            *(v4h*)(buf + vrow * 32 + vpart) = vals;
        } else {
            // ---- elementwise fallback (CX=3 head chunk, CH=16 tail, zeros)
#pragma unroll
            for (int i = 0; i < 4; ++i) {
                int e  = threadIdx.x + i * 128;
                int kk = e & 31;
                int k  = ks * 32 + kk;
                float v = 0.0f;
                if (k < KAP) {
                    if (KAR == KAP || k < KAR) {
                        int dy = k / (3 * CX);
                        int r  = k - dy * (3 * CX);
                        int dx = r / CX;
                        int c  = r - dx * CX;
                        int iy = sy0[i] * S + dy;
                        int ix = sx0[i] * S + dx;
                        if (iy < IH && ix < IW)
                            v = (float)srcX[bX[i] + (iy * IW + ix) * CX + c];
                    }
                } else if (KTOT == KP || k < KTOT) {
                    if (srcH) {
                        int kh = k - KAP;
                        int dy = kh / (3 * CH);
                        int r  = kh - dy * (3 * CH);
                        int dx = r / CH;
                        int c  = r - dx * CH;
                        int iy = sy0[i] + dy - 1;
                        int ix = sx0[i] + dx - 1;
                        if (iy >= 0 && iy < HH && ix >= 0 && ix < HW)
                            v = (float)srcH[bH[i] + (iy * HW + ix) * CH + c];
                    }
                }
                buf[(e >> 5) * 32 + kk] = (f16)v;
            }
        }
    };

    // A fragment (16x32): lane<16 -> row=lane, K{0..7,16..23};
    //                     lane>=16 -> row=lane-16, K{8..15,24..31}.
    const int arow = ((lane & 15) << 5) + ((lane >> 4) << 3);
    // B fragment (32x16): lanes 0..15 hold K{0..7,16..23}, lanes 16..31 K{8..15,24..31}.
    const f16* btRow[NT];
#pragma unroll
    for (int j = 0; j < NT; ++j)
        btRow[j] = Bt + (size_t)(n0 + j * 16 + (lane & 15)) * KP + ((lane >> 4) << 3);

    v8f acc[NT] = {};
    stage(0, aT[0]);
#pragma unroll 2
    for (int ks = 0; ks < NKS; ++ks) {
        __syncthreads();                        // stage(ks) visible; buf[(ks+1)&1] safe to reuse
        if (ks + 1 < NKS) stage(ks + 1, aT[(ks + 1) & 1]);
        if (ks + 2 < NKS) __builtin_prefetch(btRow[0] + (ks + 2) * 32, 0, 3);
        const f16* buf = aT[ks & 1];
        V16H a;
        a.h[0] = *(const v8h*)(buf + arow);
        a.h[1] = *(const v8h*)(buf + arow + 16);
#pragma unroll
        for (int j = 0; j < NT; ++j) {
            V16H b;
            b.h[0] = *(const v8h*)(btRow[j] + ks * 32);
            b.h[1] = *(const v8h*)(btRow[j] + ks * 32 + 16);
            acc[j] = __builtin_amdgcn_wmma_f32_16x16x32_f16(false, a.v, false, b.v,
                                                            (short)0, acc[j], false, false);
        }
    }

    // D layout: VGPR r, lanes 0..15 -> M = r, lanes 16..31 -> M = 8+r.
    const int rbase = m0 + ((lane >> 4) << 3);
#pragma unroll
    for (int j = 0; j < NT; ++j) {
        int col  = n0 + j * 16 + (lane & 15);
        float bv = bias[col];
#pragma unroll
        for (int r = 0; r < 8; ++r)
            z[(size_t)(rbase + r) * N + col] = acc[j][r] + bv;
    }
}

// ---------------------------------------------------------------------------
// Elementwise LSTM gates (Keras order i,f,c,o; hard_sigmoid + relu).
// ---------------------------------------------------------------------------
template<int F, int SPAT, int TT>
__global__ void k_gates(const float* __restrict__ z, const float* __restrict__ cPrev,
                        float* __restrict__ cNew, f16* __restrict__ h,
                        f16* __restrict__ hSeq, int M, int t) {
    int i = blockIdx.x * blockDim.x + threadIdx.x;
    if (i >= M * F) return;
    int m = i / F, f = i - m * F;
    constexpr int N = 4 * F;
    const float* zr = z + (size_t)m * N;
    float ig = hsig(zr[f]);
    float fg = hsig(zr[F + f]);
    float zc = zr[2 * F + f];
    float og = hsig(zr[3 * F + f]);
    float co = cPrev ? cPrev[i] : 0.0f;
    float cn = fg * co + ig * fmaxf(zc, 0.0f);
    float hn = og * fmaxf(cn, 0.0f);
    cNew[i] = cn;
    f16 hh = (f16)hn;
    h[i] = hh;
    int b = m / SPAT, rem = m - b * SPAT;
    hSeq[((size_t)(b * TT + t) * SPAT + rem) * F + f] = hh;
}

// ---------------------------------------------------------------------------
// 2x2 spatial max pool, stride 2, SAME; compile-time geometry.
// ---------------------------------------------------------------------------
template<int H, int W, int C, int HO, int WO>
__global__ void k_maxpool(const f16* __restrict__ in, f16* __restrict__ out, int BT) {
    int i = blockIdx.x * blockDim.x + threadIdx.x;
    if (i >= BT * HO * WO * C) return;
    int c = i % C; int r = i / C;
    int wo = r % WO; r /= WO;
    int ho = r % HO; r /= HO;                    // r = bt index
    float m = -1e30f;
#pragma unroll
    for (int dy = 0; dy < 2; ++dy) {
        int y = ho * 2 + dy; if (y >= H) continue;
#pragma unroll
        for (int dx = 0; dx < 2; ++dx) {
            int x = wo * 2 + dx; if (x >= W) continue;
            m = fmaxf(m, (float)in[(((size_t)r * H + y) * W + x) * C + c]);
        }
    }
    out[i] = (f16)m;
}

// ---------------------------------------------------------------------------
// Dense GEMM: [16][K] f16 (row-major, stride lda) x Bt [N][K] f16 -> f16 out.
// One block; wave w owns columns [16w, 16w+16). K compile-time.
// ---------------------------------------------------------------------------
template<int K, int N>
__global__ __launch_bounds__(256) void k_dense(const f16* __restrict__ A, int lda,
                                               const f16* __restrict__ Bt,
                                               const float* __restrict__ bias,
                                               int applyRelu, f16* __restrict__ out) {
    const int lane = threadIdx.x & 31;
    const int wave = threadIdx.x >> 5;
    const int n0 = wave * 16;
    const f16* aRow = A  + (size_t)(lane & 15) * lda + ((lane >> 4) << 3);
    const f16* bRow = Bt + (size_t)(n0 + (lane & 15)) * K + ((lane >> 4) << 3);
    v8f acc = {};
#pragma unroll 2
    for (int k0 = 0; k0 < K; k0 += 32) {
        V16H a, b;
        a.h[0] = *(const v8h*)(aRow + k0);
        a.h[1] = *(const v8h*)(aRow + k0 + 16);
        b.h[0] = *(const v8h*)(bRow + k0);
        b.h[1] = *(const v8h*)(bRow + k0 + 16);
        acc = __builtin_amdgcn_wmma_f32_16x16x32_f16(false, a.v, false, b.v,
                                                     (short)0, acc, false, false);
    }
    const int col = n0 + (lane & 15);
    const float bv = bias[col];
    const int r0 = (lane >> 4) << 3;
#pragma unroll
    for (int r = 0; r < 8; ++r) {
        float v = acc[r] + bv;
        if (applyRelu) v = fmaxf(v, 0.0f);
        out[(r0 + r) * N + col] = (f16)v;
    }
}

// ---------------------------------------------------------------------------
// Heads: act2 [16][64] x BtH [16(cols)][64] -> logits (16,7) + value (16,1).
// d_out = logits flat (112 floats) then value flat (16 floats).
// ---------------------------------------------------------------------------
__global__ void k_head(const f16* __restrict__ act2, const f16* __restrict__ BtH,
                       const float* __restrict__ bl, const float* __restrict__ bv,
                       float* __restrict__ outAll) {
    const int lane = threadIdx.x & 31;
    const f16* aRow = act2 + (lane & 15) * 64 + ((lane >> 4) << 3);
    const f16* bRow = BtH  + (lane & 15) * 64 + ((lane >> 4) << 3);
    v8f acc = {};
#pragma unroll
    for (int k0 = 0; k0 < 64; k0 += 32) {
        V16H a, b;
        a.h[0] = *(const v8h*)(aRow + k0);
        a.h[1] = *(const v8h*)(aRow + k0 + 16);
        b.h[0] = *(const v8h*)(bRow + k0);
        b.h[1] = *(const v8h*)(bRow + k0 + 16);
        acc = __builtin_amdgcn_wmma_f32_16x16x32_f16(false, a.v, false, b.v,
                                                     (short)0, acc, false, false);
    }
    const int col = lane & 15;
    const int r0 = (lane >> 4) << 3;
#pragma unroll
    for (int r = 0; r < 8; ++r) {
        int row = r0 + r;
        if (col < 7)       outAll[row * 7 + col] = acc[r] + bl[col];
        else if (col == 7) outAll[112 + row]     = acc[r] + bv[0];
    }
}

// ---------------------------------------------------------------------------
extern "C" void kernel_launch(void* const* d_in, const int* in_sizes, int n_in,
                              void* d_out, int out_size, void* d_ws, size_t ws_size,
                              hipStream_t stream) {
    (void)in_sizes; (void)n_in; (void)out_size; (void)ws_size;
    const float* x   = (const float*)d_in[0];
    const float* k1  = (const float*)d_in[1];
    const float* rk1 = (const float*)d_in[2];
    const float* b1  = (const float*)d_in[3];
    const float* k2  = (const float*)d_in[4];
    const float* rk2 = (const float*)d_in[5];
    const float* b2  = (const float*)d_in[6];
    const float* w1  = (const float*)d_in[7];
    const float* bd1 = (const float*)d_in[8];
    const float* w2  = (const float*)d_in[9];
    const float* bd2 = (const float*)d_in[10];
    const float* wv  = (const float*)d_in[11];
    const float* bvv = (const float*)d_in[12];
    const float* wl  = (const float*)d_in[13];
    const float* bl  = (const float*)d_in[14];
    float* out = (float*)d_out;

    const int B = 16, T = 10;
    char* p = (char*)d_ws;
    auto alloc = [&](size_t bytes) -> char* {
        char* r = p;
        p += (bytes + 255) & ~(size_t)255;
        return r;
    };
    f16*   xh   = (f16*)  alloc((size_t)B * T * 96 * 96 * 3 * sizeof(f16));
    f16*   bt1  = (f16*)  alloc((size_t)256 * 608 * sizeof(f16));
    f16*   bt2  = (f16*)  alloc((size_t)64 * 736 * sizeof(f16));
    f16*   btw1 = (f16*)  alloc((size_t)128 * 5760 * sizeof(f16));
    f16*   btw2 = (f16*)  alloc((size_t)64 * 128 * sizeof(f16));
    f16*   bth  = (f16*)  alloc((size_t)16 * 64 * sizeof(f16));
    float* z1   = (float*)alloc((size_t)35344 * 256 * sizeof(float));
    float* c1   = (float*)alloc((size_t)35344 * 64 * sizeof(float));
    f16*   h1   = (f16*)  alloc((size_t)35344 * 64 * sizeof(f16));
    f16*   h1s  = (f16*)  alloc((size_t)B * T * 47 * 47 * 64 * sizeof(f16));
    f16*   p1   = (f16*)  alloc((size_t)B * T * 24 * 24 * 64 * sizeof(f16));
    float* z2   = (float*)alloc((size_t)1936 * 64 * sizeof(float));
    float* c2   = (float*)alloc((size_t)1936 * 16 * sizeof(float));
    f16*   h2   = (f16*)  alloc((size_t)1936 * 16 * sizeof(f16));
    f16*   h2s  = (f16*)  alloc((size_t)B * T * 11 * 11 * 16 * sizeof(f16));
    f16*   p2   = (f16*)  alloc((size_t)B * T * 6 * 6 * 16 * sizeof(f16));
    f16*   a1   = (f16*)  alloc((size_t)16 * 128 * sizeof(f16));
    f16*   a2   = (f16*)  alloc((size_t)16 * 64 * sizeof(f16));

    // 1) x -> f16
    {
        int n = B * T * 96 * 96 * 3;
        k_f32_to_f16<<<(n + 255) / 256, 256, 0, stream>>>(x, xh, n);
    }
    // 2) weight prep: [input kernel (K-padded to 32-aligned) ; recurrent kernel]
    {
        int n;
        n = 256 * 608;  k_prep_bt<<<(n + 255) / 256, 256, 0, stream>>>(k1, 27,  rk1, 32,  576, 256, 608, bt1);
        n = 64 * 736;   k_prep_bt<<<(n + 255) / 256, 256, 0, stream>>>(k2, 576, rk2, 576, 144, 64, 736, bt2);
        n = 128 * 5760; k_prep_bt<<<(n + 255) / 256, 256, 0, stream>>>(w1, 5760, nullptr, 5760, 0, 128, 5760, btw1);
        n = 64 * 128;   k_prep_bt<<<(n + 255) / 256, 256, 0, stream>>>(w2, 128, nullptr, 128, 0, 64, 128, btw2);
        k_prep_head<<<4, 256, 0, stream>>>(wl, wv, bth);
    }
    // 3) ConvLSTM layer 1: M = 35344 (2209 tiles), N = 256, NT=2 -> grid.y=2
    {
        dim3 grid(35344 / 16, 2);
        for (int t = 0; t < T; ++t) {
            k_convlstm_gemm<10, 3, 2, 96, 96, 64, 47, 47, 47, 47,
                            256, 27, 32, 608, 608, 2>
                <<<grid, 128, 0, stream>>>(xh, t, t ? h1 : nullptr, bt1, b1, z1);
            int ne = 35344 * 64;
            k_gates<64, 2209, 10><<<(ne + 255) / 256, 256, 0, stream>>>(
                z1, t ? c1 : nullptr, c1, h1, h1s, 35344, t);
        }
    }
    // 4) pool1: (16,10,47,47,64) -> (16,10,24,24,64)
    {
        int n = B * T * 24 * 24 * 64;
        k_maxpool<47, 47, 64, 24, 24><<<(n + 255) / 256, 256, 0, stream>>>(h1s, p1, B * T);
    }
    // 5) ConvLSTM layer 2: M = 1936 (121 tiles), N = 64, NT=1 -> grid.y=1
    {
        dim3 grid(1936 / 16, 1);
        for (int t = 0; t < T; ++t) {
            k_convlstm_gemm<10, 64, 2, 24, 24, 16, 11, 11, 11, 11,
                            64, 576, 576, 720, 736, 1>
                <<<grid, 128, 0, stream>>>(p1, t, t ? h2 : nullptr, bt2, b2, z2);
            int ne = 1936 * 16;
            k_gates<16, 121, 10><<<(ne + 255) / 256, 256, 0, stream>>>(
                z2, t ? c2 : nullptr, c2, h2, h2s, 1936, t);
        }
    }
    // 6) pool2: (16,10,11,11,16) -> (16,10,6,6,16) => flat (16,5760)
    {
        int n = B * T * 6 * 6 * 16;
        k_maxpool<11, 11, 16, 6, 6><<<(n + 255) / 256, 256, 0, stream>>>(h2s, p2, B * T);
    }
    // 7) dense 5760->128 relu; 8) 128->64 relu; 9) heads
    k_dense<5760, 128><<<1, 256, 0, stream>>>(p2, 5760, btw1, bd1, 1, a1);
    k_dense<128, 64><<<1, 128, 0, stream>>>(a1, 128, btw2, bd2, 1, a2);
    k_head<<<1, 32, 0, stream>>>(a2, bth, bl, bvv, out);
}
